// NEMON_MM_Net_73735998537938
// MI455X (gfx1250) — compile-verified, matrix operand order checked by reading, not run on hardware
//
#include <hip/hip_runtime.h>
#include <hip/hip_bf16.h>

// ---------------------------------------------------------------------------
// NEMON mixed-monotone fixed point on MI455X (gfx1250).
// Compute-bound (AI ~1.3 kFLOP/byte vs HBM) -> f16 WMMA (16x16x32, f32 acc).
// All iteration state lives in LDS for all 50 iterations (ping-pong f16),
// so HBM traffic is just x (51MB) + out (20MB).
// BT=64 rows/WG; W, Wp, Wn all resident in LDS (z needs 1 WMMA per k-step).
// ---------------------------------------------------------------------------

typedef __attribute__((ext_vector_type(16))) _Float16 v16h;
typedef __attribute__((ext_vector_type(8)))  _Float16 v8h;
typedef __attribute__((ext_vector_type(8)))  float    v8f;
typedef __attribute__((ext_vector_type(4)))  float    v4f;

#define WMMA_F16(a, b, c) \
  __builtin_amdgcn_wmma_f32_16x16x32_f16(false, (a), false, (b), (short)0, (c), false, false)

constexpr int BT    = 64;    // batch rows per workgroup
constexpr int OUTD  = 100;   // out_dim
constexpr int NPAD  = 112;   // out_dim padded to 7x16
constexpr int KPAD  = 128;   // out_dim padded to 4x32 (K of iteration GEMMs)
constexpr int IND   = 784;   // in_dim
constexpr int WST   = 136;   // f16 row stride (128 + 8 halfs -> 4-bank row skew)
constexpr int IST   = 114;   // f32 injection row stride (112 + 2)
constexpr int ITERS = 50;
constexpr int NTILE = 84;    // 3 states x 4 m-tiles x 7 n-tiles
constexpr int TPW   = 11;    // padded tiles per wave (88 total)
constexpr float ALPHA  = 0.5f;
constexpr float MARGIN = 0.1f;

static __device__ __forceinline__ v8f f8z() {
  v8f r;
#pragma unroll
  for (int i = 0; i < 8; ++i) r[i] = 0.f;
  return r;
}

// Load one 16-half fragment chunk-pair: 8 halfs at p, 8 halfs at p+16.
// Matches A(16x32)/B(32x16) f16 WMMA lane layout when caller computes
// row = l16, kb = k0 + (lane>=16 ? 8 : 0).
static __device__ __forceinline__ v16h ld16(const _Float16* p) {
  v8h c0 = *(const v8h*)p;
  v8h c1 = *(const v8h*)(p + 16);
  v16h r;
#pragma unroll
  for (int i = 0; i < 8; ++i) { r[i] = c0[i]; r[8 + i] = c1[i]; }
  return r;
}

// ------------------------- global min/max reduction -------------------------
__global__ void nemon_minmax_part(const float* __restrict__ x, int n,
                                  float* __restrict__ red) {
  __shared__ float smx[256], smn[256];
  const int tid = threadIdx.x;
  float mx = -3.402823e38f, mn = 3.402823e38f;
  for (int i = blockIdx.x * 256 + tid; i < n; i += gridDim.x * 256) {
    float v = x[i];
    mx = fmaxf(mx, v);
    mn = fminf(mn, v);
  }
  smx[tid] = mx; smn[tid] = mn;
  __syncthreads();
  for (int s = 128; s > 0; s >>= 1) {
    if (tid < s) {
      smx[tid] = fmaxf(smx[tid], smx[tid + s]);
      smn[tid] = fminf(smn[tid], smn[tid + s]);
    }
    __syncthreads();
  }
  if (tid == 0) {
    red[2 + blockIdx.x]       = smx[0];
    red[2 + 256 + blockIdx.x] = smn[0];
  }
}

__global__ void nemon_minmax_final(float* __restrict__ red) {
  __shared__ float smx[256], smn[256];
  const int tid = threadIdx.x;
  smx[tid] = red[2 + tid];
  smn[tid] = red[2 + 256 + tid];
  __syncthreads();
  for (int s = 128; s > 0; s >>= 1) {
    if (tid < s) {
      smx[tid] = fmaxf(smx[tid], smx[tid + s]);
      smn[tid] = fminf(smn[tid], smn[tid + s]);
    }
    __syncthreads();
  }
  if (tid == 0) { red[0] = smx[0]; red[1] = smn[0]; }
}

// ------------------------------- main kernel -------------------------------
__global__ __launch_bounds__(256, 1)
void nemon_mm_net_main(const float* __restrict__ x, const float* __restrict__ epsp,
                       const float* __restrict__ A, const float* __restrict__ U,
                       const float* __restrict__ b, const float* __restrict__ red,
                       float* __restrict__ out, int Bn) {
  __shared__ __align__(16) _Float16 sW [NPAD * WST];        // 30464 B
  __shared__ __align__(16) _Float16 sWp[NPAD * WST];        // 30464 B
  __shared__ __align__(16) _Float16 sWn[NPAD * WST];        // 30464 B
  __shared__ __align__(16) _Float16 sZ[2][3][BT * WST];     // 104448 B (ping-pong)
  __shared__ __align__(16) float    sInj[3][BT * IST];      // 87552 B
  __shared__ float sT[NPAD];                                //   448 B

  const int tid  = threadIdx.x;
  const int lane = tid & 31;
  const int wid  = tid >> 5;         // 0..7
  const int l16  = lane & 15;
  const int hi8  = (lane >> 4) << 3; // 0 or 8
  const int m0   = blockIdx.x * BT;

  const float eps  = epsp[0];
  const float xmax = red[0];
  const float xmin = red[1];

  // ---- row sums of |offdiag(A)| and zero state buffers ----
  for (int n = tid; n < NPAD; n += 256) {
    float t = 0.f;
    if (n < OUTD) {
      for (int j = 0; j < OUTD; ++j)
        if (j != n) t += fabsf(A[n * OUTD + j]);
    }
    sT[n] = t;
  }
  {
    _Float16* p = &sZ[0][0][0];
    for (int i = tid; i < 2 * 3 * BT * WST; i += 256) p[i] = (_Float16)0.f;
  }
  __syncthreads();

  // ---- build W / Wp / Wn (f16, zero-padded to 112x128) ----
  for (int idx = tid; idx < NPAD * KPAD; idx += 256) {
    const int r = idx >> 7, c = idx & 127;
    float v = 0.f;
    if (r < OUTD && c < OUTD)
      v = (r == c) ? ((1.0f - MARGIN) - sT[r]) : A[r * OUTD + c];
    sW [r * WST + c] = (_Float16)v;
    sWp[r * WST + c] = (_Float16)fmaxf(v, 0.f);
    sWn[r * WST + c] = (_Float16)fminf(v, 0.f);
  }

  // ---- injection GEMMs: u = xU^T+b, uh = xh Up^T + xl Un^T + b, ul = ... ----
  // 28 tile positions (4 m-tiles x 7 n-tiles), t = i*8+wid, positions-outer so
  // only 3 accumulators are live at a time.
#pragma unroll 1
  for (int i = 0; i < 4; ++i) {
    const int t = i * 8 + wid;
    if (t >= 28) continue;                      // wave-uniform dummy
    const int mt = t / 7, nt = t % 7;
    const int nrow = nt * 16 + l16;             // U row (column of U^T)
    const bool urowok = (nrow < OUTD);
    const float* xp = x + (size_t)(m0 + mt * 16 + l16) * IND;
    const float* up = U + (size_t)nrow * IND;

    v8f aU = f8z(), aH = f8z(), aL = f8z();

    for (int ks = 0; ks < 25; ++ks) {           // K=784 -> 25 steps (last masked)
      const int  kb = ks * 32 + hi8;
      const bool c2 = (ks != 24);               // second 8-half chunk in-bounds?

      // A-fragments: x, xh=min(x+eps,xmax), xl=max(x-eps,xmin); pads exactly 0.
      v16h fx, fxh, fxl;
      {
        v4f a0 = *(const v4f*)(xp + kb);
        v4f a1 = *(const v4f*)(xp + kb + 4);
#pragma unroll
        for (int e = 0; e < 8; ++e) {
          const float v = (e < 4) ? a0[e] : a1[e - 4];
          fx[e]  = (_Float16)v;
          fxh[e] = (_Float16)fminf(v + eps, xmax);
          fxl[e] = (_Float16)fmaxf(v - eps, xmin);
        }
        if (c2) {
          v4f b0 = *(const v4f*)(xp + kb + 16);
          v4f b1 = *(const v4f*)(xp + kb + 20);
#pragma unroll
          for (int e = 0; e < 8; ++e) {
            const float v = (e < 4) ? b0[e] : b1[e - 4];
            fx[8 + e]  = (_Float16)v;
            fxh[8 + e] = (_Float16)fminf(v + eps, xmax);
            fxl[8 + e] = (_Float16)fmaxf(v - eps, xmin);
          }
        } else {
#pragma unroll
          for (int e = 0; e < 8; ++e) {
            fx[8 + e] = (_Float16)0.f; fxh[8 + e] = (_Float16)0.f; fxl[8 + e] = (_Float16)0.f;
          }
        }
      }

      // B-fragments from U with pos/neg split (rows >= 100 are zero padding)
      v16h fU, fUp, fUn;
      {
        float v0[8], v1[8];
#pragma unroll
        for (int e = 0; e < 8; ++e) { v0[e] = 0.f; v1[e] = 0.f; }
        if (urowok) {
          v4f a0 = *(const v4f*)(up + kb);
          v4f a1 = *(const v4f*)(up + kb + 4);
#pragma unroll
          for (int e = 0; e < 8; ++e) v0[e] = (e < 4) ? a0[e] : a1[e - 4];
          if (c2) {
            v4f b0 = *(const v4f*)(up + kb + 16);
            v4f b1 = *(const v4f*)(up + kb + 20);
#pragma unroll
            for (int e = 0; e < 8; ++e) v1[e] = (e < 4) ? b0[e] : b1[e - 4];
          }
        }
#pragma unroll
        for (int e = 0; e < 8; ++e) {
          const float t0 = v0[e], t1 = v1[e];
          fU[e]  = (_Float16)t0;             fU[8 + e]  = (_Float16)t1;
          fUp[e] = (_Float16)fmaxf(t0, 0.f); fUp[8 + e] = (_Float16)fmaxf(t1, 0.f);
          fUn[e] = (_Float16)fminf(t0, 0.f); fUn[8 + e] = (_Float16)fminf(t1, 0.f);
        }
      }

      aU = WMMA_F16(fx,  fU,  aU);
      aH = WMMA_F16(fxh, fUp, aH);
      aH = WMMA_F16(fxl, fUn, aH);
      aL = WMMA_F16(fxl, fUp, aL);
      aL = WMMA_F16(fxh, fUn, aL);
    }

    // store injections (+b) to LDS f32
    const int ncol = nt * 16 + l16;
    const float bn = (ncol < OUTD) ? b[ncol] : 0.f;
#pragma unroll
    for (int r = 0; r < 8; ++r) {
      const int mrow = mt * 16 + r + hi8;
      sInj[0][mrow * IST + ncol] = aU[r] + bn;
      sInj[1][mrow * IST + ncol] = aH[r] + bn;
      sInj[2][mrow * IST + ncol] = aL[r] + bn;
    }
  }
  __syncthreads();

  // ---- 50 averaged Picard iterations, all in LDS ----
  // 84 real tiles (3 states x 4 m-tiles x 7 n-tiles), t = i*8+wid, 11 per wave.
  // s=0 (z): 1 WMMA per k-step with full W; s=1/2: 2 WMMAs with Wp/Wn.
#pragma unroll 1
  for (int it = 0; it < ITERS; ++it) {
    const int cur = it & 1, nxt = cur ^ 1;
    v8f acc[TPW];
#pragma unroll
    for (int i = 0; i < TPW; ++i) acc[i] = f8z();

    for (int ks = 0; ks < 4; ++ks) {           // K = 128
      const int kb = ks * 32 + hi8;
#pragma unroll
      for (int i = 0; i < TPW; ++i) {
        const int t = i * 8 + wid;
        if (t >= NTILE) continue;              // wave-uniform dummy
        const int s   = t / 28;
        const int rem = t % 28;
        const int mt  = rem / 7;
        const int nt  = rem % 7;
        const int arow = mt * 16 + l16;
        const int brow = nt * 16 + l16;
        if (s == 0) {
          // z W^T
          v16h fa = ld16(&sZ[cur][0][0] + arow * WST + kb);
          v16h w  = ld16(sW + brow * WST + kb);
          acc[i] = WMMA_F16(fa, w, acc[i]);
        } else {
          // s=1: zh Wp^T + zl Wn^T ; s=2: zl Wp^T + zh Wn^T
          const int sa = (s == 1) ? 1 : 2;
          const int sb = (s == 1) ? 2 : 1;
          v16h fa = ld16(&sZ[cur][sa][0] + arow * WST + kb);
          v16h fb = ld16(&sZ[cur][sb][0] + arow * WST + kb);
          v16h wp = ld16(sWp + brow * WST + kb);
          v16h wn = ld16(sWn + brow * WST + kb);
          acc[i] = WMMA_F16(fa, wp, acc[i]);
          acc[i] = WMMA_F16(fb, wn, acc[i]);
        }
      }
    }

    // update: z_new = (1-a) z + a relu(acc + inj); write to ping-pong buffer
#pragma unroll
    for (int i = 0; i < TPW; ++i) {
      const int t = i * 8 + wid;
      if (t >= NTILE) continue;
      const int s   = t / 28;
      const int rem = t % 28;
      const int mt  = rem / 7;
      const int nt  = rem % 7;
      const int ncol = nt * 16 + l16;
      const _Float16* oldp = &sZ[cur][s][0];
      _Float16*       newp = &sZ[nxt][s][0];
#pragma unroll
      for (int r = 0; r < 8; ++r) {
        const int mrow = mt * 16 + r + hi8;
        const float zo = (float)oldp[mrow * WST + ncol];
        const float zn = fmaxf(acc[i][r] + sInj[s][mrow * IST + ncol], 0.f);
        newp[mrow * WST + ncol] = (_Float16)((1.f - ALPHA) * zo + ALPHA * zn);
      }
    }
    __syncthreads();
  }

  // ---- write [3, B, 100] output ----
  const int fin = ITERS & 1;  // final ping-pong buffer
  for (int idx = tid; idx < 3 * BT * OUTD; idx += 256) {
    const int s   = idx / (BT * OUTD);
    const int rem = idx % (BT * OUTD);
    const int r   = rem / OUTD;
    const int n   = rem % OUTD;
    out[(size_t)s * Bn * OUTD + (size_t)(m0 + r) * OUTD + n] =
        (float)sZ[fin][s][r * WST + n];
  }
}

// ------------------------------- launcher ----------------------------------
extern "C" void kernel_launch(void* const* d_in, const int* in_sizes, int n_in,
                              void* d_out, int out_size, void* d_ws, size_t ws_size,
                              hipStream_t stream) {
  (void)n_in; (void)out_size; (void)ws_size;
  const float* x   = (const float*)d_in[0];
  const float* eps = (const float*)d_in[1];
  const float* A   = (const float*)d_in[2];
  const float* U   = (const float*)d_in[3];
  const float* b   = (const float*)d_in[4];
  float* out = (float*)d_out;
  float* red = (float*)d_ws;  // [0]=xmax, [1]=xmin, [2..257]=max parts, [258..513]=min parts

  const int nx = in_sizes[0];
  const int Bn = nx / IND;

  nemon_minmax_part<<<256, 256, 0, stream>>>(x, nx, red);
  nemon_minmax_final<<<1, 256, 0, stream>>>(red);
  nemon_mm_net_main<<<Bn / BT, 256, 0, stream>>>(x, eps, A, U, b, red, out, Bn);
}
